// PoseGraphPredictionLayer_85100482003175
// MI455X (gfx1250) — matrix operand order repaired and checked by reading, not compiled, not run on hardware
//
#include <hip/hip_runtime.h>

// ---------------------------------------------------------------------------
// PoseGraphPrediction GraphNet layer for gfx1250 (MI455X, wave32, WMMA).
// Strategy: f16 WMMA (v_wmma_f32_16x16x32_f16) with f32 accumulation for all
// matmuls; LDS-staged activation tiles; weights pre-transposed/padded to f16.
// ---------------------------------------------------------------------------

typedef __attribute__((ext_vector_type(16))) _Float16 v16h;
typedef __attribute__((ext_vector_type(8)))  _Float16 v8h;
typedef __attribute__((ext_vector_type(4)))  _Float16 v4h;
typedef __attribute__((ext_vector_type(8)))  float    v8f;

#define WAVES 8            // 256 threads / 32
#define TILE_ROWS 64       // rows (edges/nodes) per workgroup

// LDS row strides (halves), chosen so row->row bank shift has gcd(shift,64)==4
// => conflict-free half-wave b128 A-fragment loads.
#define XE_STRIDE 232      // for K=224 input (E_IN=208 padded)
#define XN_STRIDE 168      // for K=160 input (N_IN=144 padded)
#define H_STRIDE  136      // for K=128 hidden
#define O_STRIDE  65       // f32 output rows (conflict-free column walks)

// ---------------------------------------------------------------------------
// One MLP layer on a 64-row LDS tile: H = [relu](Xin[64][K] @ W + bias)
// Wt is the f16 transposed weight in global memory: Wt[n][k], row length K.
// Wave-tiled: each of 8 waves owns (4*NOUT/16)/8 16x16 output tiles.
// ---------------------------------------------------------------------------
template<int K, int INS, int NOUT, int OUTS, bool RELU>
__device__ __forceinline__ void mlp_layer(const _Float16* __restrict__ Xin,
                                          const _Float16* __restrict__ Wt,
                                          const float*    __restrict__ bias,
                                          _Float16*       __restrict__ Hf16,
                                          float*          __restrict__ Hf32)
{
    const int lane = threadIdx.x & 31;
    const int wave = threadIdx.x >> 5;
    const int hs   = lane >> 4;      // half-wave select
    const int nn   = lane & 15;      // column within tile / A-row within tile
    constexpr int NT    = NOUT / 16;           // column tiles
    constexpr int NITER = (4 * NT) / WAVES;    // tiles per wave

    for (int it = 0; it < NITER; ++it) {
        const int t = wave + it * WAVES;
        const int r = t / NT;                  // row tile (0..3)
        const int c = t % NT;                  // col tile

        const float bb = bias[c * 16 + nn];
        v8f acc = {bb, bb, bb, bb, bb, bb, bb, bb};

        const _Float16* arow = Xin + (r * 16 + nn) * INS;  // A row for this lane
        const _Float16* wrow = Wt  + (size_t)(c * 16 + nn) * K;

        #pragma unroll
        for (int kk = 0; kk < K; kk += 32) {
            // A fragment (16x32 f16): halves 0..7 = K kk+hs*8.., 8..15 = +16
            v8h alo = *(const v8h*)(arow + kk + hs * 8);
            v8h ahi = *(const v8h*)(arow + kk + hs * 8 + 16);
            v16h a;
            #pragma unroll
            for (int i = 0; i < 8; ++i) { a[i] = alo[i]; a[i + 8] = ahi[i]; }
            // B fragment (32x16 f16): halves 0..7 = K kk+hs*8.., 8..15 = +16
            v8h blo = *(const v8h*)(wrow + kk + hs * 8);
            v8h bhi = *(const v8h*)(wrow + kk + hs * 8 + 16);
            v16h b;
            #pragma unroll
            for (int i = 0; i < 8; ++i) { b[i] = blo[i]; b[i + 8] = bhi[i]; }
            acc = __builtin_amdgcn_wmma_f32_16x16x32_f16(
                    false, a, false, b, (short)0, acc, false, false);
        }

        #pragma unroll
        for (int j = 0; j < 8; ++j) {
            const int row_o = r * 16 + hs * 8 + j;
            const int col   = c * 16 + nn;
            if constexpr (RELU) {
                float v = acc[j] > 0.0f ? acc[j] : 0.0f;
                Hf16[row_o * OUTS + col] = (_Float16)v;
            } else {
                Hf32[row_o * OUTS + col] = acc[j];
            }
        }
    }
}

// LayerNorm(64) in place on [64][O_STRIDE] f32 LDS tile; one thread per row.
__device__ __forceinline__ void ln_rows_inplace(float* buf, const float* g,
                                                const float* b)
{
    const int t = threadIdx.x;
    if (t < 64) {
        float* row = buf + t * O_STRIDE;
        float s = 0.0f, s2 = 0.0f;
        #pragma unroll
        for (int c = 0; c < 64; ++c) { float v = row[c]; s += v; s2 += v * v; }
        const float mu  = s  * 0.015625f;
        const float var = s2 * 0.015625f - mu * mu;
        const float inv = rsqrtf(var + 1e-5f);
        #pragma unroll
        for (int c = 0; c < 64; ++c)
            row[c] = (row[c] - mu) * inv * g[c] + b[c];
    }
}

// LayerNorm + sigmoid on abuf rows, multiply into ebuf (attention gate).
__device__ __forceinline__ void ln_sig_mul(const float* abuf, float* ebuf,
                                           const float* g, const float* b)
{
    const int t = threadIdx.x;
    if (t < 64) {
        const float* row  = abuf + t * O_STRIDE;
        float*       erow = ebuf + t * O_STRIDE;
        float s = 0.0f, s2 = 0.0f;
        #pragma unroll
        for (int c = 0; c < 64; ++c) { float v = row[c]; s += v; s2 += v * v; }
        const float mu  = s  * 0.015625f;
        const float var = s2 * 0.015625f - mu * mu;
        const float inv = rsqrtf(var + 1e-5f);
        #pragma unroll
        for (int c = 0; c < 64; ++c) {
            float a = (row[c] - mu) * inv * g[c] + b[c];
            erow[c] *= 1.0f / (1.0f + __expf(-a));
        }
    }
}

// ---------------------------------------------------------------------------
// Weight prep: w[K][N] f32 -> wt[N][KPAD] f16 (transposed, zero-padded K)
// ---------------------------------------------------------------------------
__global__ void conv_wt_kernel(const float* __restrict__ w,
                               _Float16* __restrict__ wt,
                               int K, int N, int KPAD)
{
    const int i = blockIdx.x * blockDim.x + threadIdx.x;
    if (i >= N * KPAD) return;
    const int n = i / KPAD;
    const int k = i - n * KPAD;
    wt[i] = (k < K) ? (_Float16)w[(size_t)k * N + n] : (_Float16)0.0f;
}

// ---------------------------------------------------------------------------
// Edge model: 64 edges per workgroup. Gather 208-wide concat -> two MLP
// chains (e + attention) -> LN, sigmoid-gate -> atomic scatter into agg.
// ---------------------------------------------------------------------------
__global__ __launch_bounds__(256) void edge_kernel(
    const float* __restrict__ x, const float* __restrict__ ea,
    const float* __restrict__ u, const int* __restrict__ ei,
    const int* __restrict__ batch, int n_edges,
    const _Float16* __restrict__ eW1, const _Float16* __restrict__ eW2,
    const _Float16* __restrict__ eW3,
    const float* __restrict__ e_b1, const float* __restrict__ e_b2,
    const float* __restrict__ e_b3, const float* __restrict__ e_g,
    const float* __restrict__ e_beta,
    const _Float16* __restrict__ aW1, const _Float16* __restrict__ aW2,
    const _Float16* __restrict__ aW3,
    const float* __restrict__ a_b1, const float* __restrict__ a_b2,
    const float* __restrict__ a_b3, const float* __restrict__ a_g,
    const float* __restrict__ a_beta,
    float* __restrict__ agg)
{
    __shared__ _Float16 Xe[TILE_ROWS * XE_STRIDE];
    __shared__ _Float16 H0[TILE_ROWS * H_STRIDE];
    __shared__ _Float16 H1[TILE_ROWS * H_STRIDE];
    __shared__ float    Eout[TILE_ROWS * O_STRIDE];
    __shared__ float    Aout[TILE_ROWS * O_STRIDE];
    __shared__ int      s_src[TILE_ROWS], s_tgt[TILE_ROWS], s_gb[TILE_ROWS];

    const int e0  = blockIdx.x * TILE_ROWS;
    const int tid = threadIdx.x;
    const int* srcp = ei;
    const int* tgtp = ei + n_edges;

    if (tid < TILE_ROWS) {
        const int ge = e0 + tid;
        if (ge < n_edges) {
            const int s = srcp[ge];
            s_src[tid] = s;
            s_tgt[tid] = tgtp[ge];
            s_gb[tid]  = batch[s];
        } else {
            s_src[tid] = 0; s_tgt[tid] = -1; s_gb[tid] = 0;
        }
    }
    __syncthreads();

    // Gather concat [x[src](64) | x[tgt](64) | edge_attr(64) | u(16) | 0(16)]
    for (int idx = tid; idx < TILE_ROWS * 56; idx += 256) {
        const int e = idx / 56;
        const int c = (idx % 56) * 4;
        float4 v = {0.0f, 0.0f, 0.0f, 0.0f};
        if (s_tgt[e] >= 0) {
            if (c < 64)       v = *(const float4*)(x + (size_t)s_src[e] * 64 + c);
            else if (c < 128) v = *(const float4*)(x + (size_t)s_tgt[e] * 64 + (c - 64));
            else if (c < 192) v = *(const float4*)(ea + (size_t)(e0 + e) * 64 + (c - 128));
            else if (c < 208) v = *(const float4*)(u + (size_t)s_gb[e] * 16 + (c - 192));
        }
        v4h h = {(_Float16)v.x, (_Float16)v.y, (_Float16)v.z, (_Float16)v.w};
        *(v4h*)(Xe + e * XE_STRIDE + c) = h;
    }
    __syncthreads();

    // --- edge feature MLP chain ---
    mlp_layer<224, XE_STRIDE, 128, H_STRIDE, true >(Xe, eW1, e_b1, H0, nullptr); __syncthreads();
    mlp_layer<128, H_STRIDE, 128, H_STRIDE, true >(H0, eW2, e_b2, H1, nullptr); __syncthreads();
    mlp_layer<128, H_STRIDE, 128, H_STRIDE, true >(H1, eW2 + 128 * 128, e_b2 + 128, H0, nullptr); __syncthreads();
    mlp_layer<128, H_STRIDE,  64, O_STRIDE, false>(H0, eW3, e_b3, nullptr, Eout); __syncthreads();
    ln_rows_inplace(Eout, e_g, e_beta);
    __syncthreads();

    // --- attention MLP chain ---
    mlp_layer<224, XE_STRIDE, 128, H_STRIDE, true >(Xe, aW1, a_b1, H0, nullptr); __syncthreads();
    mlp_layer<128, H_STRIDE, 128, H_STRIDE, true >(H0, aW2, a_b2, H1, nullptr); __syncthreads();
    mlp_layer<128, H_STRIDE, 128, H_STRIDE, true >(H1, aW2 + 128 * 128, a_b2 + 128, H0, nullptr); __syncthreads();
    mlp_layer<128, H_STRIDE,  64, O_STRIDE, false>(H0, aW3, a_b3, nullptr, Aout); __syncthreads();
    ln_sig_mul(Aout, Eout, a_g, a_beta);
    __syncthreads();

    // Scatter-add gated edge outputs into agg[tgt]
    for (int i = tid; i < TILE_ROWS * 64; i += 256) {
        const int e = i >> 6;
        const int c = i & 63;
        const int tg = s_tgt[e];
        if (tg >= 0)
            atomicAdd(agg + (size_t)tg * 64 + c, Eout[e * O_STRIDE + c]);
    }
}

// ---------------------------------------------------------------------------
// Node model: 64 nodes per workgroup. Gather [x | agg | u(batch)] -> MLP ->
// LayerNorm -> global f32 output.
// ---------------------------------------------------------------------------
__global__ __launch_bounds__(256) void node_kernel(
    const float* __restrict__ x, const float* __restrict__ agg,
    const float* __restrict__ u, const int* __restrict__ batch, int n_nodes,
    const _Float16* __restrict__ nW1, const _Float16* __restrict__ nW2,
    const _Float16* __restrict__ nW3,
    const float* __restrict__ b1, const float* __restrict__ b2,
    const float* __restrict__ b3, const float* __restrict__ g,
    const float* __restrict__ beta,
    float* __restrict__ out)
{
    __shared__ _Float16 Xn[TILE_ROWS * XN_STRIDE];
    __shared__ _Float16 H0[TILE_ROWS * H_STRIDE];
    __shared__ _Float16 H1[TILE_ROWS * H_STRIDE];
    __shared__ float    Hout[TILE_ROWS * O_STRIDE];
    __shared__ int      s_gb[TILE_ROWS];

    const int n0  = blockIdx.x * TILE_ROWS;
    const int tid = threadIdx.x;

    if (tid < TILE_ROWS) {
        const int gn = n0 + tid;
        s_gb[tid] = (gn < n_nodes) ? batch[gn] : -1;
    }
    __syncthreads();

    // Gather concat [x(64) | agg(64) | u(16) | 0(16)]
    for (int idx = tid; idx < TILE_ROWS * 40; idx += 256) {
        const int e = idx / 40;
        const int c = (idx % 40) * 4;
        float4 v = {0.0f, 0.0f, 0.0f, 0.0f};
        const int gn = n0 + e;
        if (s_gb[e] >= 0) {
            if (c < 64)       v = *(const float4*)(x   + (size_t)gn * 64 + c);
            else if (c < 128) v = *(const float4*)(agg + (size_t)gn * 64 + (c - 64));
            else if (c < 144) v = *(const float4*)(u   + (size_t)s_gb[e] * 16 + (c - 128));
        }
        v4h h = {(_Float16)v.x, (_Float16)v.y, (_Float16)v.z, (_Float16)v.w};
        *(v4h*)(Xn + e * XN_STRIDE + c) = h;
    }
    __syncthreads();

    mlp_layer<160, XN_STRIDE, 128, H_STRIDE, true >(Xn, nW1, b1, H0, nullptr); __syncthreads();
    mlp_layer<128, H_STRIDE, 128, H_STRIDE, true >(H0, nW2, b2, H1, nullptr); __syncthreads();
    mlp_layer<128, H_STRIDE, 128, H_STRIDE, true >(H1, nW2 + 128 * 128, b2 + 128, H0, nullptr); __syncthreads();
    mlp_layer<128, H_STRIDE,  64, O_STRIDE, false>(H0, nW3, b3, nullptr, Hout); __syncthreads();
    ln_rows_inplace(Hout, g, beta);
    __syncthreads();

    for (int i = tid; i < TILE_ROWS * 64; i += 256) {
        const int e = i >> 6;
        const int c = i & 63;
        const int gn = n0 + e;
        if (gn < n_nodes)
            out[(size_t)gn * 64 + c] = Hout[e * O_STRIDE + c];
    }
}

// ---------------------------------------------------------------------------
// Host launcher
// ---------------------------------------------------------------------------
extern "C" void kernel_launch(void* const* d_in, const int* in_sizes, int n_in,
                              void* d_out, int out_size, void* d_ws, size_t ws_size,
                              hipStream_t stream)
{
    (void)n_in; (void)out_size; (void)ws_size;

    const float* x         = (const float*)d_in[0];
    const float* edge_attr = (const float*)d_in[1];
    const float* u         = (const float*)d_in[2];
    const int*   edge_idx  = (const int*)d_in[3];
    const int*   batch     = (const int*)d_in[4];

    // MLP params per prefix: w_in, b_in, w_h, b_h, w_out, b_out, ln_g, ln_b
    const float* const* P = (const float* const*)(d_in + 5);
    const float* e_p[8]; const float* a_p[8]; const float* n_p[8];
    for (int i = 0; i < 8; ++i) { e_p[i] = P[i]; a_p[i] = P[8 + i]; n_p[i] = P[16 + i]; }

    const int n_nodes = in_sizes[0] / 64;
    const int n_edges = in_sizes[3] / 2;

    // Workspace carve-out (256B aligned)
    size_t off = 0;
    auto carve = [&](size_t bytes) -> void* {
        void* p = (char*)d_ws + off;
        off += (bytes + 255) & ~(size_t)255;
        return p;
    };
    float*     agg = (float*)   carve((size_t)n_nodes * 64 * sizeof(float));
    _Float16*  eW1 = (_Float16*)carve(128 * 224 * sizeof(_Float16));
    _Float16*  eW2 = (_Float16*)carve(2 * 128 * 128 * sizeof(_Float16));
    _Float16*  eW3 = (_Float16*)carve(64 * 128 * sizeof(_Float16));
    _Float16*  aW1 = (_Float16*)carve(128 * 224 * sizeof(_Float16));
    _Float16*  aW2 = (_Float16*)carve(2 * 128 * 128 * sizeof(_Float16));
    _Float16*  aW3 = (_Float16*)carve(64 * 128 * sizeof(_Float16));
    _Float16*  nW1 = (_Float16*)carve(128 * 160 * sizeof(_Float16));
    _Float16*  nW2 = (_Float16*)carve(2 * 128 * 128 * sizeof(_Float16));
    _Float16*  nW3 = (_Float16*)carve(64 * 128 * sizeof(_Float16));

    hipMemsetAsync(agg, 0, (size_t)n_nodes * 64 * sizeof(float), stream);

    // Weight prep: transpose + f16 + K-pad
    auto conv = [&](const float* w, _Float16* wt, int K, int N, int KPAD) {
        const int total = N * KPAD;
        conv_wt_kernel<<<(total + 255) / 256, 256, 0, stream>>>(w, wt, K, N, KPAD);
    };
    conv(e_p[0], eW1, 208, 128, 224);
    conv(e_p[2],             eW2,             128, 128, 128);
    conv(e_p[2] + 128 * 128, eW2 + 128 * 128, 128, 128, 128);
    conv(e_p[4], eW3, 128, 64, 128);
    conv(a_p[0], aW1, 208, 128, 224);
    conv(a_p[2],             aW2,             128, 128, 128);
    conv(a_p[2] + 128 * 128, aW2 + 128 * 128, 128, 128, 128);
    conv(a_p[4], aW3, 128, 64, 128);
    conv(n_p[0], nW1, 144, 128, 160);
    conv(n_p[2],             nW2,             128, 128, 128);
    conv(n_p[2] + 128 * 128, nW2 + 128 * 128, 128, 128, 128);
    conv(n_p[4], nW3, 128, 64, 128);

    // Edge model + scatter
    const int eb = (n_edges + TILE_ROWS - 1) / TILE_ROWS;
    edge_kernel<<<eb, 256, 0, stream>>>(
        x, edge_attr, u, edge_idx, batch, n_edges,
        eW1, eW2, eW3, e_p[1], e_p[3], e_p[5], e_p[6], e_p[7],
        aW1, aW2, aW3, a_p[1], a_p[3], a_p[5], a_p[6], a_p[7],
        agg);

    // Node model
    const int nb = (n_nodes + TILE_ROWS - 1) / TILE_ROWS;
    node_kernel<<<nb, 256, 0, stream>>>(
        x, agg, u, batch, n_nodes,
        nW1, nW2, nW3, n_p[1], n_p[3], n_p[5], n_p[6], n_p[7],
        (float*)d_out);
}
